// SymLSTMCell_84550726189464
// MI455X (gfx1250) — matrix operand order repaired and checked by reading, not compiled, not run on hardware
//
#include <hip/hip_runtime.h>
#include <hip/hip_bf16.h>
#include <math.h>

// ---------------------------------------------------------------------------
// SymLSTMCell fused kernels for gfx1250 (MI455X).
//   gates = input @ W_ih + h0 @ W_hh + (bias_ih + bias_hh)   [8192 x 10240]
//   per hidden block of 512: c1 = tanh(g)*sig(i) + c0*sig(f); h1 = tanh(c1)*sig(o)
//
// Fast path (needs ~189MB workspace):
//   1) convert input/h0 -> bf16, transpose-convert W_ih/W_hh -> bf16 [col][k]
//   2) bf16 WMMA GEMM with async global->LDS tile copies (ASYNCcnt) and
//      fused LSTM epilogue. Zero conversion VALU in the hot loop.
// Fallback (small ws): single fused kernel converting fp32->bf16 on the fly.
// ---------------------------------------------------------------------------

typedef __attribute__((ext_vector_type(16))) __bf16          v16bf;
typedef __attribute__((ext_vector_type(8)))  float           v8f;
typedef __attribute__((ext_vector_type(8)))  unsigned short  us8;
typedef __attribute__((ext_vector_type(4)))  int             v4i;

#define BATCH   8192
#define HID     2560
#define GATE    10240
#define KD      2560
#define BM      128          // batch rows per workgroup
#define BNH     64           // hidden cols per workgroup (=> 256 gate cols)
#define BK      32           // K-step per WMMA stage
#define KT_PHASE (KD / BK)   // 80 K-tiles per phase (input phase, h0 phase)
#define KT_TOTAL (2 * KT_PHASE)
#define LDA     40           // LDS row stride (ushort) for A: 32 + 8 pad
#define LDB     40           // LDS row stride (ushort) for B (K-transposed)
#define BCOLS   256          // gate columns per workgroup tile

#if defined(__has_builtin)
#if __has_builtin(__builtin_amdgcn_global_load_async_to_lds_b128)
#define ASYNC_LDS 1
#endif
#endif

#ifdef ASYNC_LDS
typedef __attribute__((address_space(1))) v4i* gv4p;   // global int4*
typedef __attribute__((address_space(3))) v4i* lv4p;   // LDS    int4*
#endif

__device__ __forceinline__ unsigned short f2bf(float f) {
    unsigned int u = __float_as_uint(f);
    u += 0x7FFFu + ((u >> 16) & 1u);          // round-to-nearest-even
    return (unsigned short)(u >> 16);
}
__device__ __forceinline__ float sigm(float x) { return 1.0f / (1.0f + __expf(-x)); }

union FragU { us8 h[2]; v16bf v; };

// 16-byte global->LDS copy: async DMA on CDNA5, plain load+ds_store otherwise.
__device__ __forceinline__ void copy16(const unsigned short* g, unsigned short* l) {
#ifdef ASYNC_LDS
    __builtin_amdgcn_global_load_async_to_lds_b128((gv4p)g, (lv4p)l, 0, 0);
#else
    *(us8*)l = *(const us8*)g;
#endif
}
__device__ __forceinline__ void wait_async_tile(bool one_tile_pending) {
#ifdef ASYNC_LDS
    if (one_tile_pending) __builtin_amdgcn_s_wait_asynccnt(6);  // 6 ops/thread/tile
    else                  __builtin_amdgcn_s_wait_asynccnt(0);
#endif
}

// ---- shared LSTM epilogue: acc[mt][gate][nt] -> h1/c1 ----------------------
__device__ __forceinline__ void lstm_epilogue(
    v8f (&acc)[2][4][2], int lane, int waveM, int waveN,
    int bRow, int hBase, long gateBase,
    const float* __restrict__ c0, const float* __restrict__ bih,
    const float* __restrict__ bhh, float* __restrict__ out)
{
    const int nLane = lane & 15;
    float bias[4][2];
#pragma unroll
    for (int g = 0; g < 4; ++g)
#pragma unroll
        for (int nt = 0; nt < 2; ++nt) {
            const long c = gateBase + (long)g * 512 + waveN * 32 + nt * 16 + nLane;
            bias[g][nt] = bih[c] + bhh[c];
        }
    float* hOut = out;
    float* cOut = out + (size_t)BATCH * HID;
#pragma unroll
    for (int mt = 0; mt < 2; ++mt) {
#pragma unroll
        for (int nt = 0; nt < 2; ++nt) {
            const int H = hBase + waveN * 32 + nt * 16 + nLane;
#pragma unroll
            for (int r = 0; r < 8; ++r) {
                const int m = bRow + waveM * 32 + mt * 16 + (lane >> 4) * 8 + r;
                const size_t off = (size_t)m * HID + H;
                const float gv = acc[mt][0][nt][r] + bias[0][nt];
                const float iv = acc[mt][1][nt][r] + bias[1][nt];
                const float fv = acc[mt][2][nt][r] + bias[2][nt];
                const float ov = acc[mt][3][nt][r] + bias[3][nt];
                const float c1 = tanhf(gv) * sigm(iv) + c0[off] * sigm(fv);
                hOut[off] = tanhf(c1) * sigm(ov);
                cOut[off] = c1;
            }
        }
    }
}

// ===========================================================================
// Pre-pass 1: elementwise fp32 -> bf16 (4 elems / thread)
// ===========================================================================
__global__ __launch_bounds__(256)
void cvt_f32_bf16_kernel(const float* __restrict__ in, unsigned short* __restrict__ out) {
    const size_t i = ((size_t)blockIdx.x * blockDim.x + threadIdx.x) * 4;
    const float4 v = *(const float4*)(in + i);
    const unsigned int p0 = (unsigned)f2bf(v.x) | ((unsigned)f2bf(v.y) << 16);
    const unsigned int p1 = (unsigned)f2bf(v.z) | ((unsigned)f2bf(v.w) << 16);
    *(uint2*)(out + i) = make_uint2(p0, p1);
}

// ===========================================================================
// Pre-pass 2: W[k][col] fp32 -> Wt[col][k] bf16 (32x32 LDS tile transpose)
// ===========================================================================
__global__ __launch_bounds__(256)
void transpose_cvt_kernel(const float* __restrict__ W, unsigned short* __restrict__ Wt) {
    __shared__ unsigned short t[32][33];
    const int c0 = blockIdx.x * 32;
    const int k0 = blockIdx.y * 32;
    const int tc = threadIdx.x & 31;
    const int tr = threadIdx.x >> 5;   // 8 rows per pass
#pragma unroll
    for (int rr = 0; rr < 4; ++rr) {
        const int k = tr + rr * 8;
        t[k][tc] = f2bf(W[(size_t)(k0 + k) * GATE + c0 + tc]);
    }
    __syncthreads();
#pragma unroll
    for (int rr = 0; rr < 4; ++rr) {
        const int c = tr + rr * 8;
        Wt[(size_t)(c0 + c) * KD + k0 + tc] = t[tc][c];
    }
}

// ===========================================================================
// Main GEMM (fast path): bf16 sources, async LDS tile copies, fused epilogue
// ===========================================================================
__global__ __launch_bounds__(256)
void symlstm_wmma_bf16_kernel(const unsigned short* __restrict__ aB,    // input  bf16 [8192][2560]
                              const unsigned short* __restrict__ hB,    // h0     bf16 [8192][2560]
                              const unsigned short* __restrict__ wihT,  // W_ih^T bf16 [10240][2560]
                              const unsigned short* __restrict__ whhT,  // W_hh^T bf16 [10240][2560]
                              const float* __restrict__ c0,
                              const float* __restrict__ bih,
                              const float* __restrict__ bhh,
                              float* __restrict__ out)
{
    __shared__ unsigned short sA[2][BM * LDA];     // 20 KB
    __shared__ unsigned short sB[2][BCOLS * LDB];  // 40 KB ([col][k])

    const int tid   = threadIdx.x;
    const int lane  = tid & 31;
    const int wave  = tid >> 5;
    const int waveM = wave & 3;
    const int waveN = wave >> 2;

    const int  bRow  = blockIdx.y * BM;
    const int  hBase = blockIdx.x * BNH;
    const int  blk   = hBase / 512;
    const int  j0    = hBase % 512;
    const long gateBase = (long)blk * 2048 + j0;

    // ---- per-thread 16-byte copy coordinates (2 for A, 4 for B) ----
    size_t aGOff[2]; int aLOff[2];
#pragma unroll
    for (int i = 0; i < 2; ++i) {
        const int idx = tid + i * 256;           // 512 x b128 covers 128x32 bf16
        const int row = idx >> 2, q = idx & 3;
        aGOff[i] = (size_t)(bRow + row) * KD + q * 8;
        aLOff[i] = row * LDA + q * 8;
    }
    size_t bGOff[4]; int bLOff[4];
#pragma unroll
    for (int i = 0; i < 4; ++i) {
        const int idx = tid + i * 256;           // 1024 x b128 covers 256x32 bf16
        const int col = idx >> 2, q = idx & 3;
        const long gcol = gateBase + (long)(col >> 6) * 512 + (col & 63);
        bGOff[i] = (size_t)gcol * KD + q * 8;
        bLOff[i] = col * LDB + q * 8;
    }

    auto issueTile = [&](int kt, int buf) {
        const unsigned short* Ap = (kt < KT_PHASE) ? aB : hB;
        const unsigned short* Wp = (kt < KT_PHASE) ? wihT : whhT;
        const int k0 = (kt % KT_PHASE) * BK;
#pragma unroll
        for (int i = 0; i < 2; ++i) copy16(Ap + aGOff[i] + k0, &sA[buf][aLOff[i]]);
#pragma unroll
        for (int i = 0; i < 4; ++i) copy16(Wp + bGOff[i] + k0, &sB[buf][bLOff[i]]);
    };

    // ---- accumulators ----
    v8f acc[2][4][2];
#pragma unroll
    for (int mt = 0; mt < 2; ++mt)
#pragma unroll
        for (int g = 0; g < 4; ++g)
#pragma unroll
            for (int nt = 0; nt < 2; ++nt) {
                v8f z = {0.f, 0.f, 0.f, 0.f, 0.f, 0.f, 0.f, 0.f};
                acc[mt][g][nt] = z;
            }

    // WMMA fragment coordinates (ISA 7.12.2, wave32)
    const int aKLo  = (lane >> 4) * 8;
    const int aRowL = waveM * 32 + (lane & 15);
    const int bKLo  = (lane >> 4) * 8;
    const int nLane = lane & 15;

    issueTile(0, 0);
    issueTile(1, 1);

    for (int kt = 0; kt < KT_TOTAL; ++kt) {
        const int cur = kt & 1;
        wait_async_tile((kt + 1) < KT_TOTAL);  // tile kt complete (<=1 tile in flight)
        __syncthreads();

        FragU af0, af1;
        af0.h[0] = *(const us8*)&sA[cur][(aRowL + 0 ) * LDA + aKLo];
        af0.h[1] = *(const us8*)&sA[cur][(aRowL + 0 ) * LDA + 16 + aKLo];
        af1.h[0] = *(const us8*)&sA[cur][(aRowL + 16) * LDA + aKLo];
        af1.h[1] = *(const us8*)&sA[cur][(aRowL + 16) * LDA + 16 + aKLo];

#pragma unroll
        for (int g = 0; g < 4; ++g) {
#pragma unroll
            for (int nt = 0; nt < 2; ++nt) {
                const int col = g * 64 + waveN * 32 + nt * 16 + nLane;
                FragU bf;
                bf.h[0] = *(const us8*)&sB[cur][col * LDB + bKLo];
                bf.h[1] = *(const us8*)&sB[cur][col * LDB + bKLo + 16];
                acc[0][g][nt] = __builtin_amdgcn_wmma_f32_16x16x32_bf16(
                    false, af0.v, false, bf.v, (short)0, acc[0][g][nt], false, false);
                acc[1][g][nt] = __builtin_amdgcn_wmma_f32_16x16x32_bf16(
                    false, af1.v, false, bf.v, (short)0, acc[1][g][nt], false, false);
            }
        }

        __syncthreads();                       // everyone done reading buf[cur]
        if (kt + 2 < KT_TOTAL) issueTile(kt + 2, cur);
    }

    lstm_epilogue(acc, lane, waveM, waveN, bRow, hBase, gateBase, c0, bih, bhh, out);
}

// ===========================================================================
// Fallback: fused fp32->bf16 conversion inside the GEMM (no workspace needed)
// ===========================================================================
__global__ __launch_bounds__(256)
void symlstm_wmma_fused_kernel(const float* __restrict__ input,
                               const float* __restrict__ h0,
                               const float* __restrict__ c0,
                               const float* __restrict__ wih,
                               const float* __restrict__ whh,
                               const float* __restrict__ bih,
                               const float* __restrict__ bhh,
                               float* __restrict__ out)
{
    __shared__ unsigned short sA[2][BM * LDA];
    __shared__ unsigned short sB[2][BCOLS * LDB];

    const int tid   = threadIdx.x;
    const int lane  = tid & 31;
    const int wave  = tid >> 5;
    const int waveM = wave & 3;
    const int waveN = wave >> 2;

    const int  bRow  = blockIdx.y * BM;
    const int  hBase = blockIdx.x * BNH;
    const int  blk   = hBase / 512;
    const int  j0    = hBase % 512;
    const long gateBase = (long)blk * 2048 + j0;

    int aRow[4], aQ[4];
#pragma unroll
    for (int i = 0; i < 4; ++i) { int idx = tid + i * 256; aRow[i] = idx >> 3; aQ[i] = idx & 7; }
    int bK[8], bG[8], bF[8];
#pragma unroll
    for (int i = 0; i < 8; ++i) { int idx = tid + i * 256; bK[i] = idx >> 6; int r = idx & 63; bG[i] = r >> 4; bF[i] = r & 15; }

    float4 ra[4];
    float4 rb[8];

    auto loadGlobal = [&](int kt) {
        const float* Ap = (kt < KT_PHASE) ? input : h0;
        const float* Wp = (kt < KT_PHASE) ? wih   : whh;
        const int k0 = (kt % KT_PHASE) * BK;
#pragma unroll
        for (int i = 0; i < 4; ++i)
            ra[i] = *(const float4*)(Ap + (size_t)(bRow + aRow[i]) * KD + k0 + aQ[i] * 4);
#pragma unroll
        for (int i = 0; i < 8; ++i)
            rb[i] = *(const float4*)(Wp + (size_t)(k0 + bK[i]) * GATE + gateBase + (long)bG[i] * 512 + bF[i] * 4);
    };
    auto storeLDS = [&](int buf) {
#pragma unroll
        for (int i = 0; i < 4; ++i) {
            unsigned int p0 = (unsigned)f2bf(ra[i].x) | ((unsigned)f2bf(ra[i].y) << 16);
            unsigned int p1 = (unsigned)f2bf(ra[i].z) | ((unsigned)f2bf(ra[i].w) << 16);
            *(uint2*)&sA[buf][aRow[i] * LDA + aQ[i] * 4] = make_uint2(p0, p1);
        }
#pragma unroll
        for (int i = 0; i < 8; ++i) {
            const int cb = bG[i] * 64 + bF[i] * 4;
            const int k  = bK[i];
            sB[buf][(cb + 0) * LDB + k] = f2bf(rb[i].x);
            sB[buf][(cb + 1) * LDB + k] = f2bf(rb[i].y);
            sB[buf][(cb + 2) * LDB + k] = f2bf(rb[i].z);
            sB[buf][(cb + 3) * LDB + k] = f2bf(rb[i].w);
        }
    };

    v8f acc[2][4][2];
#pragma unroll
    for (int mt = 0; mt < 2; ++mt)
#pragma unroll
        for (int g = 0; g < 4; ++g)
#pragma unroll
            for (int nt = 0; nt < 2; ++nt) {
                v8f z = {0.f, 0.f, 0.f, 0.f, 0.f, 0.f, 0.f, 0.f};
                acc[mt][g][nt] = z;
            }

    const int aKLo  = (lane >> 4) * 8;
    const int aRowL = waveM * 32 + (lane & 15);
    const int bKLo  = (lane >> 4) * 8;
    const int nLane = lane & 15;

    loadGlobal(0);
    storeLDS(0);
    __syncthreads();

    for (int kt = 0; kt < KT_TOTAL; ++kt) {
        const int  cur  = kt & 1;
        const bool more = (kt + 1) < KT_TOTAL;
        if (more) loadGlobal(kt + 1);

        FragU af0, af1;
        af0.h[0] = *(const us8*)&sA[cur][(aRowL + 0 ) * LDA + aKLo];
        af0.h[1] = *(const us8*)&sA[cur][(aRowL + 0 ) * LDA + 16 + aKLo];
        af1.h[0] = *(const us8*)&sA[cur][(aRowL + 16) * LDA + aKLo];
        af1.h[1] = *(const us8*)&sA[cur][(aRowL + 16) * LDA + 16 + aKLo];

#pragma unroll
        for (int g = 0; g < 4; ++g) {
#pragma unroll
            for (int nt = 0; nt < 2; ++nt) {
                const int col = g * 64 + waveN * 32 + nt * 16 + nLane;
                FragU bf;
                bf.h[0] = *(const us8*)&sB[cur][col * LDB + bKLo];
                bf.h[1] = *(const us8*)&sB[cur][col * LDB + bKLo + 16];
                acc[0][g][nt] = __builtin_amdgcn_wmma_f32_16x16x32_bf16(
                    false, af0.v, false, bf.v, (short)0, acc[0][g][nt], false, false);
                acc[1][g][nt] = __builtin_amdgcn_wmma_f32_16x16x32_bf16(
                    false, af1.v, false, bf.v, (short)0, acc[1][g][nt], false, false);
            }
        }

        if (more) {
            __syncthreads();
            storeLDS((kt + 1) & 1);
            __syncthreads();
        }
    }

    lstm_epilogue(acc, lane, waveM, waveN, bRow, hBase, gateBase, c0, bih, bhh, out);
}

// ===========================================================================
extern "C" void kernel_launch(void* const* d_in, const int* in_sizes, int n_in,
                              void* d_out, int out_size, void* d_ws, size_t ws_size,
                              hipStream_t stream) {
    (void)in_sizes; (void)n_in; (void)out_size;
    const float* input = (const float*)d_in[0];
    const float* h0    = (const float*)d_in[1];
    const float* c0    = (const float*)d_in[2];
    const float* wih   = (const float*)d_in[3];
    const float* whh   = (const float*)d_in[4];
    const float* bih   = (const float*)d_in[5];
    const float* bhh   = (const float*)d_in[6];
    float* out = (float*)d_out;

    const size_t SZ_A = (size_t)BATCH * KD;   // 20,971,520 elems
    const size_t SZ_W = (size_t)KD * GATE;    // 26,214,400 elems
    const size_t NEED = (2 * SZ_A + 2 * SZ_W) * sizeof(unsigned short);

    dim3 grid(HID / BNH, BATCH / BM);         // (40, 64)

    if (ws_size >= NEED) {
        unsigned short* inB  = (unsigned short*)d_ws;
        unsigned short* h0B  = inB  + SZ_A;
        unsigned short* wihT = h0B  + SZ_A;
        unsigned short* whhT = wihT + SZ_W;

        cvt_f32_bf16_kernel<<<(int)(SZ_A / 1024), 256, 0, stream>>>(input, inB);
        cvt_f32_bf16_kernel<<<(int)(SZ_A / 1024), 256, 0, stream>>>(h0, h0B);
        transpose_cvt_kernel<<<dim3(GATE / 32, KD / 32), 256, 0, stream>>>(wih, wihT);
        transpose_cvt_kernel<<<dim3(GATE / 32, KD / 32), 256, 0, stream>>>(whh, whhT);

        symlstm_wmma_bf16_kernel<<<grid, 256, 0, stream>>>(
            inB, h0B, wihT, whhT, c0, bih, bhh, out);
    } else {
        symlstm_wmma_fused_kernel<<<grid, 256, 0, stream>>>(
            input, h0, c0, wih, whh, bih, bhh, out);
    }
}
